// SPECTREAttention_3478923510286
// MI455X (gfx1250) — matrix operand order repaired and checked by reading, not compiled, not run on hardware
//
#include <hip/hip_runtime.h>
#include <hip/hip_bf16.h>
#include <math.h>

// ---------------------------------------------------------------------------
// SPECTRE attention, algebraically simplified:
//   ifft(fft(v)*g).real == Re(g) * v          (g is a per-head scalar)
//   mean_n(x @ Wq)      == mean_n(x) @ Wq     (so q is never materialized)
// Heavy work = 2 GEMMs [16384,1024]@[1024,1024] with v_wmma_f32_16x16x32_f16,
// fed by double-buffered global_load_async_to_lds_b128 (ASYNCcnt) staging.
// ---------------------------------------------------------------------------

typedef __attribute__((ext_vector_type(16))) _Float16 v16h;
typedef __attribute__((ext_vector_type(8)))  _Float16 v8h;
typedef __attribute__((ext_vector_type(8)))  float    v8f;

#define DIMC   1024
#define NTOK   4096
#define BATCH  4
#define MROWS  (BATCH * NTOK)   // 16384
#define BM 128
#define BN 128
#define BK 32

// ---- CDNA5 async global->LDS copy helpers (ASYNCcnt-tracked) --------------
__device__ __forceinline__ void async_ld_b128(unsigned ldsaddr, const _Float16* g) {
    asm volatile("global_load_async_to_lds_b128 %0, %1, off"
                 :: "v"(ldsaddr), "v"(g) : "memory");
}
__device__ __forceinline__ void async_ld_b128_o16(unsigned ldsaddr, const _Float16* g) {
    // INST_OFFSET is added to BOTH the LDS and global address (ISA 08 §4.4)
    asm volatile("global_load_async_to_lds_b128 %0, %1, off offset:16"
                 :: "v"(ldsaddr), "v"(g) : "memory");
}
__device__ __forceinline__ unsigned lds_addr32(const void* p) {
    // generic LDS aperture pointer: low 32 bits are the LDS byte address
    return (unsigned)(unsigned long long)p;
}

// ---------------- kernel 1: xbar[b,d] = mean_n x[b,n,d] ----------------
__global__ __launch_bounds__(256) void xbar_kernel(const float* __restrict__ x,
                                                   float* __restrict__ xbar) {
    int d = blockIdx.x * 256 + threadIdx.x;   // 0..1023
    int b = blockIdx.y;                        // 0..3
    const float* p = x + (size_t)b * NTOK * DIMC + d;
    float s = 0.f;
    for (int n = 0; n < NTOK; ++n) s += p[(size_t)n * DIMC];
    xbar[b * DIMC + d] = s * (1.0f / (float)NTOK);
}

// ---------------- kernel 2: per-(b,h) gate scalar (real part only) ------
__global__ __launch_bounds__(64) void gate_kernel(const float* __restrict__ xbar,
                                                  const float* __restrict__ Wq,
                                                  const float* __restrict__ ln_g,
                                                  const float* __restrict__ ln_b,
                                                  const float* __restrict__ W1,
                                                  const float* __restrict__ b1,
                                                  const float* __restrict__ W2,
                                                  const float* __restrict__ b2,
                                                  float* __restrict__ gates) {
    __shared__ float sm[64];
    const int bh = blockIdx.x;           // 0..63
    const int b  = bh >> 4, h = bh & 15;
    const int j  = threadIdx.x;          // 0..63
    const float* xb = xbar + b * DIMC;
    float q = 0.f;
    for (int k = 0; k < DIMC; ++k) q += xb[k] * Wq[(size_t)k * DIMC + h * 64 + j];
    sm[j] = q;
    __syncthreads();
    float mu = 0.f;
    for (int k = 0; k < 64; ++k) mu += sm[k];
    mu *= (1.f / 64.f);
    float var = 0.f;
    for (int k = 0; k < 64; ++k) { float d = sm[k] - mu; var += d * d; }
    var *= (1.f / 64.f);
    float qln = (q - mu) * rsqrtf(var + 1e-5f) * ln_g[j] + ln_b[j];
    __syncthreads();
    sm[j] = qln;
    __syncthreads();
    float acc = b1[j];
    for (int k = 0; k < 64; ++k) acc += sm[k] * W1[k * 64 + j];
    float hg = 0.5f * acc * (1.0f + erff(acc * 0.70710678118654752f));  // exact gelu
    __syncthreads();
    sm[j] = hg;
    __syncthreads();
    if (j == 0) {
        float l0 = b2[0];
        for (int k = 0; k < 64; ++k) l0 += sm[k] * W2[k * 2 + 0];
        gates[bh] = l0;   // Re(g): only component that survives .real
    }
}

// ---------------- kernel 3: f32 -> f16 bulk convert (8 elems/thread) ----
__global__ __launch_bounds__(256) void cvt_f16_kernel(const float* __restrict__ in,
                                                      _Float16* __restrict__ out) {
    size_t i = ((size_t)blockIdx.x * 256 + threadIdx.x) * 8;
    float4 f0 = *(const float4*)(in + i);
    float4 f1 = *(const float4*)(in + i + 4);
    v8h h;
    h[0] = (_Float16)f0.x; h[1] = (_Float16)f0.y;
    h[2] = (_Float16)f0.z; h[3] = (_Float16)f0.w;
    h[4] = (_Float16)f1.x; h[5] = (_Float16)f1.y;
    h[6] = (_Float16)f1.z; h[7] = (_Float16)f1.w;
    *(v8h*)(out + i) = h;
}

// ------- kernel 4: W [K][N] f32 -> WT [N][K] f16 (tiled transpose) ------
__global__ __launch_bounds__(256) void wT_kernel(const float* __restrict__ W,
                                                 _Float16* __restrict__ WT) {
    __shared__ float t[32][33];
    const int tx = threadIdx.x & 31, ty = threadIdx.x >> 5;  // 32 x 8
    const int k0 = blockIdx.y * 32, n0 = blockIdx.x * 32;
#pragma unroll
    for (int i = 0; i < 4; ++i) {
        int k = ty + i * 8;
        t[k][tx] = W[(size_t)(k0 + k) * DIMC + n0 + tx];
    }
    __syncthreads();
#pragma unroll
    for (int i = 0; i < 4; ++i) {
        int n = ty + i * 8;
        WT[(size_t)(n0 + n) * DIMC + k0 + tx] = (_Float16)t[tx][n];
    }
}

// ---------------- WMMA GEMM: C[M,N] = Ah[M,K] @ Bt[N,K]^T ---------------
// Ah: f16 row-major [M,K].  Bt: f16 TRANSPOSED weights [N,K].
// OUT_SCALED_F16: store f16 scaled by gates[b*16 + n/64] (v-proj pass),
//                 else plain f32 store (out-proj pass).
template <bool OUT_SCALED_F16>
__global__ __launch_bounds__(256) void gemm_wmma(const _Float16* __restrict__ Ah,
                                                 const _Float16* __restrict__ Bt,
                                                 void* __restrict__ Cptr,
                                                 const float* __restrict__ gates,
                                                 int M, int K, int N) {
    __shared__ alignas(32) _Float16 As[2][BM * BK];  // [m][k], 8 KB x2
    __shared__ alignas(32) _Float16 Bs[2][BN * BK];  // [n][k], 8 KB x2

    const int tid   = threadIdx.x;
    const int lane  = tid & 31;
    const int wave  = tid >> 5;      // 0..7
    const int waveM = wave & 3;      // 4 waves along M -> 32 rows each
    const int waveN = wave >> 2;     // 2 waves along N -> 64 cols each
    const int m0 = blockIdx.y * BM;
    const int n0 = blockIdx.x * BN;

    const int lm     = lane & 15;
    const int aKbase = (lane < 16) ? 0 : 16;  // A frag: contiguous 16 K vals
    const int bKbase = (lane < 16) ? 0 : 16;  // B frag: contiguous 16 K vals

    v8f acc[2][4] = {};

    // staging map: thread copies 32B of A row (ar) and 32B of Bt row (ar)
    const int ar = tid >> 1;             // 0..127
    const int ac = (tid & 1) * 16;       // 0 or 16 (f16 elems)
    const _Float16* ga = Ah + (size_t)(m0 + ar) * K + ac;
    const _Float16* gb = Bt + (size_t)(n0 + ar) * K + ac;
    const unsigned aL[2] = { lds_addr32(&As[0][ar * BK + ac]),
                             lds_addr32(&As[1][ar * BK + ac]) };
    const unsigned bL[2] = { lds_addr32(&Bs[0][ar * BK + ac]),
                             lds_addr32(&Bs[1][ar * BK + ac]) };

    auto issue = [&](int buf, int kElem) {   // 4 async ops per wave
        async_ld_b128    (aL[buf], ga + kElem);
        async_ld_b128_o16(aL[buf], ga + kElem);
        async_ld_b128    (bL[buf], gb + kElem);
        async_ld_b128_o16(bL[buf], gb + kElem);
    };

    const int NT = K / BK;
    issue(0, 0);
    for (int it = 0; it < NT; ++it) {
        const int buf = it & 1;
        if (it + 1 < NT) {
            issue(buf ^ 1, (it + 1) * BK);
            asm volatile("s_wait_asynccnt 4" ::: "memory");  // current tile landed
        } else {
            asm volatile("s_wait_asynccnt 0" ::: "memory");
        }
        __syncthreads();

        v16h afrag[2];
#pragma unroll
        for (int mt = 0; mt < 2; ++mt) {
            int m = waveM * 32 + mt * 16 + lm;
            v8h lo = *(const v8h*)(&As[buf][m * BK + aKbase]);
            v8h hi = *(const v8h*)(&As[buf][m * BK + aKbase + 8]);
            afrag[mt] = __builtin_shufflevector(lo, hi, 0, 1, 2, 3, 4, 5, 6, 7,
                                                8, 9, 10, 11, 12, 13, 14, 15);
        }
#pragma unroll
        for (int nt = 0; nt < 4; ++nt) {
            int n = waveN * 64 + nt * 16 + lm;
            v16h bfrag = *(const v16h*)(&Bs[buf][n * BK + bKbase]);
#pragma unroll
            for (int mt = 0; mt < 2; ++mt)
                acc[mt][nt] = __builtin_amdgcn_wmma_f32_16x16x32_f16(
                    false, afrag[mt], false, bfrag, (short)0, acc[mt][nt],
                    false, false);
        }
        __syncthreads();   // all waves done reading buf before it is re-filled
    }

    // ---- epilogue: C 16x16 f32 layout: VGPR r -> lanes0-15 M=r, 16-31 M=r+8
#pragma unroll
    for (int mt = 0; mt < 2; ++mt) {
#pragma unroll
        for (int nt = 0; nt < 4; ++nt) {
            int mBase = m0 + waveM * 32 + mt * 16 + ((lane < 16) ? 0 : 8);
            int nCol  = n0 + waveN * 64 + nt * 16 + lm;
#pragma unroll
            for (int r = 0; r < 8; ++r) {
                int   mRow = mBase + r;
                float val  = acc[mt][nt][r];
                if constexpr (OUT_SCALED_F16) {
                    // b = mRow/4096, head = nCol/64 -> Re(gate) scaling
                    float s = gates[((mRow >> 12) << 4) + (nCol >> 6)];
                    ((_Float16*)Cptr)[(size_t)mRow * N + nCol] = (_Float16)(val * s);
                } else {
                    ((float*)Cptr)[(size_t)mRow * N + nCol] = val;
                }
            }
        }
    }
}

// ---------------------------------------------------------------------------
extern "C" void kernel_launch(void* const* d_in, const int* in_sizes, int n_in,
                              void* d_out, int out_size, void* d_ws, size_t ws_size,
                              hipStream_t stream) {
    const float* x    = (const float*)d_in[0];
    const float* Wq   = (const float*)d_in[1];
    const float* Wv   = (const float*)d_in[2];
    const float* Wo   = (const float*)d_in[3];
    const float* ln_g = (const float*)d_in[4];
    const float* ln_b = (const float*)d_in[5];
    const float* gW1  = (const float*)d_in[6];
    const float* gb1  = (const float*)d_in[7];
    const float* gW2  = (const float*)d_in[8];
    const float* gb2  = (const float*)d_in[9];
    float* out = (float*)d_out;

    // ws layout: xh(32MB) | vs(32MB) | WvT(2MB) | WoT(2MB) | xbar | gates
    char*     ws    = (char*)d_ws;
    _Float16* xh    = (_Float16*)ws;
    _Float16* vs    = (_Float16*)(ws + (size_t)MROWS * DIMC * 2);
    _Float16* WvT   = (_Float16*)(ws + (size_t)MROWS * DIMC * 4);
    _Float16* WoT   = WvT + (size_t)DIMC * DIMC;
    float*    xbar  = (float*)(WoT + (size_t)DIMC * DIMC);
    float*    gates = xbar + BATCH * DIMC;

    // gate path (tiny)
    xbar_kernel<<<dim3(DIMC / 256, BATCH), 256, 0, stream>>>(x, xbar);
    gate_kernel<<<64, 64, 0, stream>>>(xbar, Wq, ln_g, ln_b, gW1, gb1, gW2, gb2, gates);

    // one-time precision/layout conversion
    cvt_f16_kernel<<<(MROWS * DIMC) / (256 * 8), 256, 0, stream>>>(x, xh);
    wT_kernel<<<dim3(32, 32), 256, 0, stream>>>(Wv, WvT);
    wT_kernel<<<dim3(32, 32), 256, 0, stream>>>(Wo, WoT);

    dim3 grid(DIMC / BN, MROWS / BM);  // (8, 128)
    // vs = (x @ Wv) * Re(g)   [f16 out, gate-scaled]
    gemm_wmma<true><<<grid, 256, 0, stream>>>(xh, WvT, vs, gates, MROWS, DIMC, DIMC);
    // out = vs @ Wo           [f32 out]
    gemm_wmma<false><<<grid, 256, 0, stream>>>(vs, WoT, out, nullptr, MROWS, DIMC, DIMC);
}
